// FeedForward_27650999451939
// MI455X (gfx1250) — matrix operand order repaired and checked by reading, compiled, not run on hardware
//
#include <hip/hip_runtime.h>
#include <cstdint>
#include <cstddef>

// ---------------------------------------------------------------------------
// Quantized SwiGLU MLP for MI455X (gfx1250, wave32, WMMA).
//   h   = silu((x @ (Wg-128)^T) * gs) * ((x @ (Wu-128)^T) * us)      [f16 in ws]
//   out = (h @ (Wd-128)^T) * ds                                      [f32]
// int8-valued weights are exactly representable in f16, so the GEMMs run
// f16 WMMA with f32 accumulation; per-channel scales applied in epilogues.
//
// Preferred path (ws large enough): one-time cvt kernels produce f16 copies
// of x and all weights in ws; both GEMMs then stream every operand via
// gfx1250 async global->LDS DMA (ASYNCcnt) -> ds_load_b128 frags -> v_wmma,
// with ~zero VALU in the hot loop. The f16 working set (~181 MB) fits the
// 192 MB L2. Fallback path converts during staging (VALU-heavier).
// ---------------------------------------------------------------------------

typedef __attribute__((ext_vector_type(16))) _Float16 v16h;
typedef __attribute__((ext_vector_type(8)))  _Float16 v8h;
typedef __attribute__((ext_vector_type(8)))  float    v8f;
typedef __attribute__((ext_vector_type(4)))  float    v4f;
typedef __attribute__((ext_vector_type(4)))  int      v4i;

#define WMMA16(a, b, c) \
  __builtin_amdgcn_wmma_f32_16x16x32_f16(false, (a), false, (b), (short)0, (c), false, false)

constexpr int BK  = 32;   // K step (matches 16x16x32 WMMA K)
constexpr int LDK = 40;   // padded LDS row stride (halves) to spread banks

// ---- gfx1250 async global->LDS (ASYNCcnt path) ----------------------------
// ISA 10.2: flat addresses in the LDS aperture carry the LDS byte offset in
// their low 32 bits (aperture base low bits are zero), so truncation is valid.
__device__ __forceinline__ uint32_t lds_addr32(const void* p) {
  return (uint32_t)(uintptr_t)p;
}
__device__ __forceinline__ void async_g2l_b128(const void* gptr, void* lptr) {
  asm volatile("global_load_async_to_lds_b128 %0, %1, off"
               :: "v"(lds_addr32(lptr)), "v"(gptr)
               : "memory");
}
__device__ __forceinline__ void wait_async0() {
  asm volatile("s_wait_asynccnt 0x0" ::: "memory");
}

// ---- WMMA fragment loads from LDS (layouts per CDNA5 ISA) ------
// A 16x32 f16: lane l (row = l&15): K 0..7,16..23 (lanes 0-15) / 8..15,24..31
__device__ __forceinline__ v16h frag_A16(const _Float16* t, unsigned lane) {
  const v8h* p = (const v8h*)(t + (lane & 15u) * LDK);
  unsigned s = (lane >> 4) & 1u;
  v8h lo = p[s];
  v8h hi = p[2 + s];
  return __builtin_shufflevector(lo, hi, 0,1,2,3,4,5,6,7,8,9,10,11,12,13,14,15);
}
// B 32x16 f16: lane l (col = l&15): K 0..7,16..23 (lanes 0-15) / 8..15,24..31
__device__ __forceinline__ v16h frag_B16(const _Float16* t, unsigned lane) {
  const v8h* p = (const v8h*)(t + (lane & 15u) * LDK);
  unsigned s = (lane >> 4) & 1u;
  v8h lo = p[s];
  v8h hi = p[2 + s];
  return __builtin_shufflevector(lo, hi, 0,1,2,3,4,5,6,7,8,9,10,11,12,13,14,15);
}

// Fallback staging: 16 int32 weights -> 16 f16 (w-128) into LDS
__device__ __forceinline__ void stage_w16(const int* __restrict__ wp, _Float16* dst) {
  const v4i* gp = (const v4i*)wp;
  v4i w0 = gp[0], w1 = gp[1], w2 = gp[2], w3 = gp[3];
  v8h a, b;
  a[0]=(_Float16)(w0[0]-128); a[1]=(_Float16)(w0[1]-128);
  a[2]=(_Float16)(w0[2]-128); a[3]=(_Float16)(w0[3]-128);
  a[4]=(_Float16)(w1[0]-128); a[5]=(_Float16)(w1[1]-128);
  a[6]=(_Float16)(w1[2]-128); a[7]=(_Float16)(w1[3]-128);
  b[0]=(_Float16)(w2[0]-128); b[1]=(_Float16)(w2[1]-128);
  b[2]=(_Float16)(w2[2]-128); b[3]=(_Float16)(w2[3]-128);
  b[4]=(_Float16)(w3[0]-128); b[5]=(_Float16)(w3[1]-128);
  b[6]=(_Float16)(w3[2]-128); b[7]=(_Float16)(w3[3]-128);
  *(v8h*)dst       = a;
  *(v8h*)(dst + 8) = b;
}

// ===========================================================================
// One-time format conversion kernels (preferred path)
// ===========================================================================
__global__ __launch_bounds__(256) void cvt_w_kernel(
    const int* __restrict__ w, _Float16* __restrict__ o, int n8) {
  for (int i = blockIdx.x * blockDim.x + threadIdx.x; i < n8;
       i += gridDim.x * blockDim.x) {
    const v4i* p = (const v4i*)(w + (size_t)i * 8);
    v4i a = p[0], b = p[1];
    v8h r;
    r[0]=(_Float16)(a[0]-128); r[1]=(_Float16)(a[1]-128);
    r[2]=(_Float16)(a[2]-128); r[3]=(_Float16)(a[3]-128);
    r[4]=(_Float16)(b[0]-128); r[5]=(_Float16)(b[1]-128);
    r[6]=(_Float16)(b[2]-128); r[7]=(_Float16)(b[3]-128);
    *(v8h*)(o + (size_t)i * 8) = r;
  }
}
__global__ __launch_bounds__(256) void cvt_x_kernel(
    const float* __restrict__ x, _Float16* __restrict__ o, int n8) {
  for (int i = blockIdx.x * blockDim.x + threadIdx.x; i < n8;
       i += gridDim.x * blockDim.x) {
    const v4f* p = (const v4f*)(x + (size_t)i * 8);
    v4f a = p[0], b = p[1];
    v8h r;
    r[0]=(_Float16)a[0]; r[1]=(_Float16)a[1]; r[2]=(_Float16)a[2]; r[3]=(_Float16)a[3];
    r[4]=(_Float16)b[0]; r[5]=(_Float16)b[1]; r[6]=(_Float16)b[2]; r[7]=(_Float16)b[3];
    *(v8h*)(o + (size_t)i * 8) = r;
  }
}

// ===========================================================================
// Kernel 1: fused gate/up GEMM + SiLU -> h (f16)
// Block tile 64(M) x 128(I), 8 waves as 2x4, wave tile 32x32 (2x2 WMMA x2 mats)
// WF16: all operands pre-converted f16 -> async DMA staging, no VALU convert.
// ===========================================================================
template<bool WF16>
__global__ __launch_bounds__(256) void gateup_kernel(
    const void* __restrict__ xv,  const void* __restrict__ gwv,
    const float* __restrict__ gsc, const void* __restrict__ uwv,
    const float* __restrict__ usc, _Float16* __restrict__ h,
    int M, int H, int I)
{
  constexpr int BM = 64, BN = 128;
  __shared__ __align__(16) _Float16 As[2][BM][LDK];
  __shared__ __align__(16) _Float16 Gs[2][BN][LDK];
  __shared__ __align__(16) _Float16 Us[2][BN][LDK];

  const unsigned tid  = threadIdx.x;
  const unsigned lane = tid & 31u;
  const unsigned wid  = tid >> 5;
  const unsigned wm   = wid >> 2;   // 0..1
  const unsigned wn   = wid & 3u;   // 0..3
  const int m0 = blockIdx.y * BM;
  const int n0 = blockIdx.x * BN;

  v8f accg[2][2] = {};
  v8f accu[2][2] = {};

  auto stage = [&](int k, int buf) {
    if constexpr (WF16) {
      const _Float16* x16 = (const _Float16*)xv;
      const _Float16* g16 = (const _Float16*)gwv;
      const _Float16* u16 = (const _Float16*)uwv;
      { // activations 64x32 f16: one b128 per thread
        int row = tid >> 2, c = (tid & 3) << 3;
        async_g2l_b128(x16 + (size_t)(m0 + row) * H + (k + c), &As[buf][row][c]);
      }
      { // weights 128x32 f16 each: two b128 per thread per matrix
        int row = tid >> 1, c = (tid & 1) << 4;
        size_t off = (size_t)(n0 + row) * H + (k + c);
        async_g2l_b128(g16 + off,     &Gs[buf][row][c]);
        async_g2l_b128(g16 + off + 8, &Gs[buf][row][c + 8]);
        async_g2l_b128(u16 + off,     &Us[buf][row][c]);
        async_g2l_b128(u16 + off + 8, &Us[buf][row][c + 8]);
      }
    } else {
      const float* x = (const float*)xv;
      const int* gw = (const int*)gwv;
      const int* uw = (const int*)uwv;
      { // activations: fp32 -> f16 via VGPRs
        int row = tid >> 2, c = (tid & 3) << 3;
        const float* gp = x + (size_t)(m0 + row) * H + (k + c);
        v4f f0 = *(const v4f*)gp;
        v4f f1 = *(const v4f*)(gp + 4);
        v8h o;
        o[0]=(_Float16)f0[0]; o[1]=(_Float16)f0[1]; o[2]=(_Float16)f0[2]; o[3]=(_Float16)f0[3];
        o[4]=(_Float16)f1[0]; o[5]=(_Float16)f1[1]; o[6]=(_Float16)f1[2]; o[7]=(_Float16)f1[3];
        *(v8h*)&As[buf][row][c] = o;
      }
      { // weights: int32 -> f16(w-128) via VGPRs
        int row = tid >> 1, c = (tid & 1) << 4;
        size_t goff = (size_t)(n0 + row) * H + (k + c);
        stage_w16(gw + goff, &Gs[buf][row][c]);
        stage_w16(uw + goff, &Us[buf][row][c]);
      }
    }
  };

  stage(0, 0);
  wait_async0();
  __syncthreads();

  const int NK = H / BK;
  for (int kt = 0; kt < NK; ++kt) {
    const int buf = kt & 1;
    if (kt + 1 < NK) stage((kt + 1) * BK, buf ^ 1);

    v16h af[2], gf[2], uf[2];
#pragma unroll
    for (int mi = 0; mi < 2; ++mi)
      af[mi] = frag_A16(&As[buf][wm * 32 + mi * 16][0], lane);
#pragma unroll
    for (int ni = 0; ni < 2; ++ni) {
      gf[ni] = frag_B16(&Gs[buf][wn * 32 + ni * 16][0], lane);
      uf[ni] = frag_B16(&Us[buf][wn * 32 + ni * 16][0], lane);
    }
#pragma unroll
    for (int mi = 0; mi < 2; ++mi)
#pragma unroll
      for (int ni = 0; ni < 2; ++ni) {
        accg[mi][ni] = WMMA16(af[mi], gf[ni], accg[mi][ni]);
        accu[mi][ni] = WMMA16(af[mi], uf[ni], accu[mi][ni]);
      }
    wait_async0();
    __syncthreads();
  }

  // Epilogue: per-channel scale, SiLU(g)*u, store f16 h.
  // C layout: col = lane&15, row = (lane>>4)*8 + r (per VGPR r).
  float gs[2], us[2];
#pragma unroll
  for (int ni = 0; ni < 2; ++ni) {
    int col = n0 + wn * 32 + ni * 16 + (lane & 15u);
    gs[ni] = gsc[col];
    us[ni] = usc[col];
  }
#pragma unroll
  for (int mi = 0; mi < 2; ++mi)
#pragma unroll
    for (int ni = 0; ni < 2; ++ni) {
      int col = n0 + wn * 32 + ni * 16 + (lane & 15u);
#pragma unroll
      for (int r = 0; r < 8; ++r) {
        int row = m0 + wm * 32 + mi * 16 + ((lane >> 4) << 3) + r;
        float g = accg[mi][ni][r] * gs[ni];
        float u = accu[mi][ni][r] * us[ni];
        float s = g / (1.0f + __expf(-g));   // silu
        h[(size_t)row * I + col] = (_Float16)(s * u);
      }
    }
}

// ===========================================================================
// Kernel 2: down GEMM: out = (h @ (Wd-128)^T) * ds
// Block tile 128(M) x 128(H), 8 waves as 2x4, wave tile 64x32 (4x2 WMMA).
// h tile always async; weights async (WF16) or VGPR-converted (fallback).
// ===========================================================================
template<bool WF16>
__global__ __launch_bounds__(256) void down_kernel(
    const _Float16* __restrict__ h, const void* __restrict__ dwv,
    const float* __restrict__ dsc,  float* __restrict__ out,
    int M, int H, int I)
{
  constexpr int BM = 128, BN = 128;
  __shared__ __align__(16) _Float16 As[2][BM][LDK];
  __shared__ __align__(16) _Float16 Ds[2][BN][LDK];

  const unsigned tid  = threadIdx.x;
  const unsigned lane = tid & 31u;
  const unsigned wid  = tid >> 5;
  const unsigned wm   = wid >> 2;   // 0..1 -> 64 rows each
  const unsigned wn   = wid & 3u;   // 0..3 -> 32 cols each
  const int m0 = blockIdx.y * BM;
  const int n0 = blockIdx.x * BN;

  v8f acc[4][2] = {};

  auto stage = [&](int k, int buf) {
    { // h tile 128x32 f16: two async b128 per thread
      int row = tid >> 1, c = (tid & 1) << 4;
      const _Float16* gp = h + (size_t)(m0 + row) * I + (k + c);
      async_g2l_b128(gp,     &As[buf][row][c]);
      async_g2l_b128(gp + 8, &As[buf][row][c + 8]);
    }
    if constexpr (WF16) {
      const _Float16* w16 = (const _Float16*)dwv;
      int row = tid >> 1, c = (tid & 1) << 4;
      size_t off = (size_t)(n0 + row) * I + (k + c);
      async_g2l_b128(w16 + off,     &Ds[buf][row][c]);
      async_g2l_b128(w16 + off + 8, &Ds[buf][row][c + 8]);
    } else {
      const int* dw = (const int*)dwv;
      int row = tid >> 1, c = (tid & 1) << 4;
      stage_w16(dw + (size_t)(n0 + row) * I + (k + c), &Ds[buf][row][c]);
    }
  };

  stage(0, 0);
  wait_async0();
  __syncthreads();

  const int NK = I / BK;
  for (int kt = 0; kt < NK; ++kt) {
    const int buf = kt & 1;
    if (kt + 1 < NK) stage((kt + 1) * BK, buf ^ 1);

    v16h af[4], bf[2];
#pragma unroll
    for (int mi = 0; mi < 4; ++mi)
      af[mi] = frag_A16(&As[buf][wm * 64 + mi * 16][0], lane);
#pragma unroll
    for (int ni = 0; ni < 2; ++ni)
      bf[ni] = frag_B16(&Ds[buf][wn * 32 + ni * 16][0], lane);
#pragma unroll
    for (int mi = 0; mi < 4; ++mi)
#pragma unroll
      for (int ni = 0; ni < 2; ++ni)
        acc[mi][ni] = WMMA16(af[mi], bf[ni], acc[mi][ni]);

    wait_async0();
    __syncthreads();
  }

  float ds[2];
#pragma unroll
  for (int ni = 0; ni < 2; ++ni)
    ds[ni] = dsc[n0 + wn * 32 + ni * 16 + (lane & 15u)];
#pragma unroll
  for (int mi = 0; mi < 4; ++mi)
#pragma unroll
    for (int ni = 0; ni < 2; ++ni) {
      int col = n0 + wn * 32 + ni * 16 + (lane & 15u);
#pragma unroll
      for (int r = 0; r < 8; ++r) {
        int row = m0 + wm * 64 + mi * 16 + ((lane >> 4) << 3) + r;
        out[(size_t)row * H + col] = acc[mi][ni][r] * ds[ni];
      }
    }
}

// ===========================================================================
extern "C" void kernel_launch(void* const* d_in, const int* in_sizes, int n_in,
                              void* d_out, int out_size, void* d_ws, size_t ws_size,
                              hipStream_t stream) {
  const float* x   = (const float*)d_in[0];
  const int*   gw  = (const int*)  d_in[1];
  const float* gsc = (const float*)d_in[2];
  const int*   uw  = (const int*)  d_in[3];
  const float* usc = (const float*)d_in[4];
  const int*   dw  = (const int*)  d_in[5];
  const float* dsc = (const float*)d_in[6];
  float* out = (float*)d_out;

  const int I = in_sizes[2];           // up/gate out channels (8192)
  const int H = in_sizes[6];           // hidden (2048)
  const int M = in_sizes[0] / H;       // B*S tokens (4096)

  const size_t eh = (size_t)M * I;     // h elements
  const size_t ex = (size_t)M * H;     // x elements
  const size_t ew = (size_t)I * H;     // per-weight-matrix elements
  const size_t need = 2 * (eh + ex + 3 * ew);  // ~185 MB for M=4096,H=2048,I=8192

  _Float16* hbuf = (_Float16*)d_ws;    // M*I f16 (both paths)
  dim3 blk(256);
  dim3 g1(I / 128, M / 64);
  dim3 g2(H / 128, M / 128);

  if (ws_size >= need) {
    // Preferred: one-time f16 conversion, then fully async-DMA GEMMs.
    _Float16* x16 = hbuf + eh;
    _Float16* g16 = x16 + ex;
    _Float16* u16 = g16 + ew;
    _Float16* d16 = u16 + ew;
    cvt_x_kernel<<<dim3(4096), blk, 0, stream>>>(x,  x16, (int)(ex / 8));
    cvt_w_kernel<<<dim3(4096), blk, 0, stream>>>(gw, g16, (int)(ew / 8));
    cvt_w_kernel<<<dim3(4096), blk, 0, stream>>>(uw, u16, (int)(ew / 8));
    cvt_w_kernel<<<dim3(4096), blk, 0, stream>>>(dw, d16, (int)(ew / 8));
    gateup_kernel<true><<<g1, blk, 0, stream>>>(x16, g16, gsc, u16, usc, hbuf, M, H, I);
    down_kernel<true><<<g2, blk, 0, stream>>>(hbuf, d16, dsc, out, M, H, I);
  } else {
    // Fallback: convert during staging (VALU-heavier, minimal ws).
    gateup_kernel<false><<<g1, blk, 0, stream>>>(x, gw, gsc, uw, usc, hbuf, M, H, I);
    down_kernel<false><<<g2, blk, 0, stream>>>(hbuf, dw, dsc, out, M, H, I);
  }
}
